// VectorQuantizer_32418413150633
// MI455X (gfx1250) — compile-verified, matrix operand order checked by reading, not run on hardware
//
#include <hip/hip_runtime.h>

// ---------------- problem constants (from reference) ----------------
#define N_ROWS    131072          // 64 * 2048 flattened vectors
#define DIM       64              // EMBEDDING_DIM
#define K_ENT     1024            // NUM_EMBEDDINGS
#define K_TILES   64              // K_ENT / 16
#define ROW_TILES 8192            // N_ROWS / 16
#define WAVES_PER_BLOCK 8
#define BLOCK     256             // 8 wave32s
// each wave handles TWO 16-row tiles (32 rows) for B-fragment reuse
#define GRID_MAIN (ROW_TILES / (2 * WAVES_PER_BLOCK))   // 512 blocks
#define ND        8388608         // N_ROWS * DIM

// ---------------- workspace layout (bytes) ----------------
#define WS_LOSS_OFF   0           // 1 float accumulator
#define WS_CNORM_OFF  1024        // 1024 floats  (4 KB)
#define WS_PACK_OFF   8192        // packed bf16 codebook, 1024*64*2 = 128 KB

typedef __attribute__((ext_vector_type(16))) __bf16 v16bf;
typedef __attribute__((ext_vector_type(8)))  float  v8f;

union BF16x16 { v16bf v; unsigned u[8]; };
union F32x8   { v8f   v; float    f[8]; };

__device__ __forceinline__ unsigned f32_to_bf16_rne(float f) {
  unsigned u = __float_as_uint(f);
  return (u + 0x7FFFu + ((u >> 16) & 1u)) >> 16;
}
__device__ __forceinline__ float bf16_bits_to_f32(unsigned h) {
  return __uint_as_float(h << 16);
}
// Per-lane K-pair index for the 16-bit 16x32 WMMA fragment layout
// (ISA 7.12.2: VGPR v<4 holds K=2v,2v+1 (+8 for lane half), v>=4 holds K=16+...)
__device__ __forceinline__ int kmap(int v, int half) {
  return (v < 4) ? (2 * v + 8 * half) : (16 + 2 * (v - 4) + 8 * half);
}

// ---------------------------------------------------------------
// Kernel 1: pack codebook into per-lane bf16 B-fragment layout.
// pack[(tile*2+chunk)*32 + lane][v] = {bf16 c[col][k], bf16 c[col][k+1]}
// ---------------------------------------------------------------
__global__ void vq_pack_codebook(const float* __restrict__ cb,
                                 unsigned* __restrict__ pack) {
  int gid   = blockIdx.x * blockDim.x + threadIdx.x;   // 4096 threads
  int lane  = gid & 31;
  int chunk = (gid >> 5) & 1;
  int tile  = gid >> 6;
  int col   = tile * 16 + (lane & 15);
  int half  = (lane >> 4) & 1;
  unsigned* dst = pack + ((size_t)(tile * 2 + chunk) * 32 + lane) * 8;
#pragma unroll
  for (int v = 0; v < 8; ++v) {
    int k   = chunk * 32 + kmap(v, half);
    float c0 = cb[col * DIM + k];
    float c1 = cb[col * DIM + k + 1];
    dst[v] = f32_to_bf16_rne(c0) | (f32_to_bf16_rne(c1) << 16);
  }
}

// ---------------------------------------------------------------
// Kernel 2: codebook squared norms + zero the loss accumulator.
// ---------------------------------------------------------------
__global__ void vq_cnorm(const float* __restrict__ cb,
                         float* __restrict__ cnorm,
                         float* __restrict__ lossAcc) {
  int e = blockIdx.x * blockDim.x + threadIdx.x;
  if (e == 0) *lossAcc = 0.0f;
  if (e < K_ENT) {
    float s = 0.0f;
#pragma unroll 8
    for (int d = 0; d < DIM; ++d) {
      float c = cb[e * DIM + d];
      s = fmaf(c, c, s);
    }
    cnorm[e] = s;
  }
}

// ---------------------------------------------------------------
// Kernel 3: main — WMMA distance GEMM + argmin + gather + loss.
// One wave32 handles TWO 16-row tiles (32 rows) against all 1024
// entries: every B fragment loaded feeds 4 WMMAs.
// The A fragments carry y = -2x (hi/lo bf16 split, exact scaling),
// so the WMMA result is -2*x.c and the distance is a single vector
// add of ||c||^2: d2 = acc + cn -> v_pk_add_f32, no fma constant,
// no cross-accumulator SLP pairing. Running argmin stays one packed
// float per row (index in low 10 mantissa bits -> v_min3_num_f32).
// B loads are register ping-pong pipelined ahead of the WMMAs.
// ---------------------------------------------------------------

__device__ __forceinline__ void loadB(const uint4* __restrict__ pb, int t,
                                      int lane, BF16x16& b0, BF16x16& b1) {
  uint4 p = pb[(((size_t)t * 2 + 0) * 32 + lane) * 2 + 0];
  b0.u[0] = p.x; b0.u[1] = p.y; b0.u[2] = p.z; b0.u[3] = p.w;
  p = pb[(((size_t)t * 2 + 0) * 32 + lane) * 2 + 1];
  b0.u[4] = p.x; b0.u[5] = p.y; b0.u[6] = p.z; b0.u[7] = p.w;
  p = pb[(((size_t)t * 2 + 1) * 32 + lane) * 2 + 0];
  b1.u[0] = p.x; b1.u[1] = p.y; b1.u[2] = p.z; b1.u[3] = p.w;
  p = pb[(((size_t)t * 2 + 1) * 32 + lane) * 2 + 1];
  b1.u[4] = p.x; b1.u[5] = p.y; b1.u[6] = p.z; b1.u[7] = p.w;
}

// build hi/lo-split A fragments for one 16-row tile from y = -2x
__device__ __forceinline__ void buildA(const float* __restrict__ xrow, int half,
                                       BF16x16& ahi0, BF16x16& alo0,
                                       BF16x16& ahi1, BF16x16& alo1) {
#pragma unroll
  for (int v = 0; v < 8; ++v) {
    int k0 = kmap(v, half);
    float f0 = -2.0f * xrow[k0], f1 = -2.0f * xrow[k0 + 1];
    unsigned h0 = f32_to_bf16_rne(f0), h1 = f32_to_bf16_rne(f1);
    ahi0.u[v] = h0 | (h1 << 16);
    alo0.u[v] = f32_to_bf16_rne(f0 - bf16_bits_to_f32(h0)) |
                (f32_to_bf16_rne(f1 - bf16_bits_to_f32(h1)) << 16);
    float g0 = -2.0f * xrow[32 + k0], g1 = -2.0f * xrow[32 + k0 + 1];
    unsigned j0 = f32_to_bf16_rne(g0), j1 = f32_to_bf16_rne(g1);
    ahi1.u[v] = j0 | (j1 << 16);
    alo1.u[v] = f32_to_bf16_rne(g0 - bf16_bits_to_f32(j0)) |
                (f32_to_bf16_rne(g1 - bf16_bits_to_f32(j1)) << 16);
  }
}

__global__ void __launch_bounds__(BLOCK) vq_main(
    const float* __restrict__ x, const float* __restrict__ cb,
    const unsigned* __restrict__ pack, const float* __restrict__ cnorm,
    float* __restrict__ outq, int* __restrict__ outidx,
    float* __restrict__ lossAcc) {
  __shared__ float s_cnorm[K_ENT];
  for (int i = threadIdx.x; i < K_ENT; i += BLOCK) s_cnorm[i] = cnorm[i];
  __syncthreads();

  const int lane = threadIdx.x & 31;
  const int wave = threadIdx.x >> 5;
  const int half = lane >> 4;
  const int l15  = lane & 15;
  const int task = blockIdx.x * WAVES_PER_BLOCK + wave;  // 32-row task
  const int rowBase = task * 32;                          // rows [rowBase,+32)

  // ---- build A fragments for both 16-row tiles ----
  BF16x16 aAhi0, aAlo0, aAhi1, aAlo1;   // rows rowBase + l15
  BF16x16 aBhi0, aBlo0, aBhi1, aBlo1;   // rows rowBase + 16 + l15
  buildA(x + (size_t)(rowBase + l15) * DIM, half, aAhi0, aAlo0, aAhi1, aAlo1);
  buildA(x + (size_t)(rowBase + 16 + l15) * DIM, half, aBhi0, aBlo0, aBhi1, aBlo1);

  // packed (distance | index) running minima
  float bestA[8], bestB[8];
#pragma unroll
  for (int r = 0; r < 8; ++r) { bestA[r] = 3.3e38f; bestB[r] = 3.3e38f; }

  const uint4* pb = (const uint4*)pack;   // 2 x uint4 per lane per fragment

  BF16x16 bP0, bP1, bQ0, bQ1;
  loadB(pb, 0, lane, bP0, bP1);

#define VQ_COMPUTE_TILE(T, B0, B1)                                            \
  do {                                                                        \
    float cn = s_cnorm[(T) * 16 + l15];                                       \
    unsigned colbits = (unsigned)((T) * 16 + l15);                            \
    F32x8 accA, accB;                                                         \
    accA.v = (v8f){0.f, 0.f, 0.f, 0.f, 0.f, 0.f, 0.f, 0.f};                   \
    accB.v = (v8f){0.f, 0.f, 0.f, 0.f, 0.f, 0.f, 0.f, 0.f};                   \
    accA.v = __builtin_amdgcn_wmma_f32_16x16x32_bf16(false, aAhi0.v, false,   \
                 (B0).v, (short)0, accA.v, false, false);                     \
    accB.v = __builtin_amdgcn_wmma_f32_16x16x32_bf16(false, aBhi0.v, false,   \
                 (B0).v, (short)0, accB.v, false, false);                     \
    accA.v = __builtin_amdgcn_wmma_f32_16x16x32_bf16(false, aAlo0.v, false,   \
                 (B0).v, (short)0, accA.v, false, false);                     \
    accB.v = __builtin_amdgcn_wmma_f32_16x16x32_bf16(false, aBlo0.v, false,   \
                 (B0).v, (short)0, accB.v, false, false);                     \
    accA.v = __builtin_amdgcn_wmma_f32_16x16x32_bf16(false, aAhi1.v, false,   \
                 (B1).v, (short)0, accA.v, false, false);                     \
    accB.v = __builtin_amdgcn_wmma_f32_16x16x32_bf16(false, aBhi1.v, false,   \
                 (B1).v, (short)0, accB.v, false, false);                     \
    accA.v = __builtin_amdgcn_wmma_f32_16x16x32_bf16(false, aAlo1.v, false,   \
                 (B1).v, (short)0, accA.v, false, false);                     \
    accB.v = __builtin_amdgcn_wmma_f32_16x16x32_bf16(false, aBlo1.v, false,   \
                 (B1).v, (short)0, accB.v, false, false);                     \
    F32x8 dA, dB;                                                             \
    dA.v = accA.v + cn;   /* vector add -> within-acc v_pk_add_f32 */         \
    dB.v = accB.v + cn;                                                       \
    _Pragma("unroll")                                                         \
    for (int r = 0; r < 8; ++r) {                                             \
      float eA = __uint_as_float((__float_as_uint(dA.f[r]) & 0xFFFFFC00u) | colbits);\
      float eB = __uint_as_float((__float_as_uint(dB.f[r]) & 0xFFFFFC00u) | colbits);\
      bestA[r] = fminf(bestA[r], eA);                                         \
      bestB[r] = fminf(bestB[r], eB);                                         \
    }                                                                         \
  } while (0)

#pragma unroll 1
  for (int t = 0; t < K_TILES; t += 2) {
    loadB(pb, t + 1, lane, bQ0, bQ1);         // in flight during tile t math
    VQ_COMPUTE_TILE(t, bP0, bP1);
    if (t + 2 < K_TILES)
      loadB(pb, t + 2, lane, bP0, bP1);       // in flight during tile t+1 math
    VQ_COMPUTE_TILE(t + 1, bQ0, bQ1);
  }
#undef VQ_COMPUTE_TILE

  // ---- butterfly argmin across the 16 lanes of each half (wave32) ----
#pragma unroll
  for (int r = 0; r < 8; ++r) {
#pragma unroll
    for (int m = 8; m >= 1; m >>= 1) {
      bestA[r] = fminf(bestA[r], __shfl_xor(bestA[r], m, 32));
      bestB[r] = fminf(bestB[r], __shfl_xor(bestB[r], m, 32));
    }
  }

  // ---- gather winner rows, write quantized_ste (== q), accumulate loss ----
  float lsum = 0.0f;
  const int dbase = l15 * 4;     // each of 16 lanes copies 4 of 64 dims
#pragma unroll
  for (int r = 0; r < 8; ++r) {
#pragma unroll
    for (int s = 0; s < 2; ++s) {            // s=0: tile A rows, s=1: tile B
      int row = rowBase + 16 * s + r + 8 * half;  // M = r + 8*half per C layout
      int idx = (int)(__float_as_uint(s ? bestB[r] : bestA[r]) & 1023u);
      if (l15 == 0) outidx[row] = idx;
      const float4 q  = *(const float4*)(cb + (size_t)idx * DIM + dbase);
      const float4 xi = *(const float4*)(x  + (size_t)row * DIM + dbase);
      *(float4*)(outq + (size_t)row * DIM + dbase) = q;
      float d0 = q.x - xi.x, d1 = q.y - xi.y, d2 = q.z - xi.z, d3 = q.w - xi.w;
      lsum = fmaf(d0, d0, lsum); lsum = fmaf(d1, d1, lsum);
      lsum = fmaf(d2, d2, lsum); lsum = fmaf(d3, d3, lsum);
    }
  }
#pragma unroll
  for (int m = 16; m >= 1; m >>= 1) lsum += __shfl_xor(lsum, m, 32);
  if (lane == 0) atomicAdd(lossAcc, lsum);
}

// ---------------------------------------------------------------
// Kernel 4: loss = (1 + 0.25) * mean((q-x)^2)
// ---------------------------------------------------------------
__global__ void vq_finalize(const float* __restrict__ lossAcc,
                            float* __restrict__ outLoss) {
  *outLoss = 1.25f * (*lossAcc) / (float)ND;
}

extern "C" void kernel_launch(void* const* d_in, const int* in_sizes, int n_in,
                              void* d_out, int out_size, void* d_ws, size_t ws_size,
                              hipStream_t stream) {
  const float* x  = (const float*)d_in[0];   // [64,2048,64] f32
  const float* cb = (const float*)d_in[1];   // [1024,64]    f32

  char*     ws      = (char*)d_ws;
  float*    lossAcc = (float*)(ws + WS_LOSS_OFF);
  float*    cnorm   = (float*)(ws + WS_CNORM_OFF);
  unsigned* pack    = (unsigned*)(ws + WS_PACK_OFF);

  float* outq    = (float*)d_out;                 // [N_ROWS*DIM] quantized_ste
  float* outLoss = outq + (size_t)ND;             // scalar loss
  int*   outidx  = (int*)(outq + (size_t)ND + 1); // [N_ROWS] indices

  vq_pack_codebook<<<16, 256, 0, stream>>>(cb, pack);
  vq_cnorm<<<4, 256, 0, stream>>>(cb, cnorm, lossAcc);
  vq_main<<<GRID_MAIN, BLOCK, 0, stream>>>(x, cb, pack, cnorm, outq, outidx, lossAcc);
  vq_finalize<<<1, 1, 0, stream>>>(lossAcc, outLoss);
}
